// TransformerDecoderModule_54709293416726
// MI455X (gfx1250) — compile-verified, matrix operand order checked by reading, not run on hardware
//
#include <hip/hip_runtime.h>
#include <hip/hip_bf16.h>

// ---------------- problem constants ----------------
constexpr int B = 8, T = 256, S = 512, D = 2176, H = 16, HD = 136, F = 8192;
constexpr int V = 1000, L = 4;
constexpr int BT = B * T;   // 2048
constexpr int BS = B * S;   // 4096
constexpr float ATT_SCALE = 0.08574929257125442f; // 1/sqrt(136)

// ---------------- WMMA types (CDNA5 wave32) ----------------
typedef __attribute__((ext_vector_type(16))) __bf16 bf16x16;
typedef __attribute__((ext_vector_type(8)))  __bf16 bf16x8;
typedef __attribute__((ext_vector_type(8)))  float  f32x8;

// ---------------- async global->LDS path (CDNA5), guarded ----------------
#if __has_builtin(__builtin_amdgcn_global_load_async_to_lds_b128) && \
    __has_builtin(__builtin_amdgcn_s_wait_asynccnt)
#define USE_ASYNC_LDS 1
// builtin signature (from hipcc diagnostic): param0 = v4i32 addrspace(1)*, param1 = v4i32 addrspace(3)*
typedef int v4i __attribute__((vector_size(16)));
typedef __attribute__((address_space(1))) v4i g_v4i;
typedef __attribute__((address_space(3))) v4i l_v4i;
__device__ __forceinline__ void async_copy16(const unsigned short* g, unsigned short* l)
{
    __builtin_amdgcn_global_load_async_to_lds_b128(
        (g_v4i*)(unsigned long long)(const void*)g,
        (l_v4i*)(unsigned long long)(void*)l, 0, 0);
}
#else
#define USE_ASYNC_LDS 0
#endif

// ---------------- f32 -> bf16 (RNE) ----------------
__global__ __launch_bounds__(256) void cvt_bf16_kernel(
    const float* __restrict__ in, unsigned short* __restrict__ out, size_t n)
{
    size_t i = (size_t)blockIdx.x * 256 + threadIdx.x;
    size_t stride = (size_t)gridDim.x * 256;
    for (; i < n; i += stride) {
        unsigned int u = __float_as_uint(in[i]);
        out[i] = (unsigned short)((u + 0x7FFFu + ((u >> 16) & 1u)) >> 16);
    }
}

// ---------------- embedding + positional ----------------
__global__ __launch_bounds__(256) void embed_kernel(
    const int* __restrict__ tok, const float* __restrict__ emb,
    const float* __restrict__ pos, float* __restrict__ out)
{
    const int row = blockIdx.x;        // b*T + t
    const int t = row % T;
    const int vix = tok[row];
    for (int i = threadIdx.x; i < D; i += 256)
        out[(size_t)row * D + i] = emb[(size_t)vix * D + i] + pos[(size_t)t * D + i];
}

// ---------------- bf16 WMMA GEMM: C[M,N] = A[M,K] @ W[N,K]^T + bias ----------------
// block tile 128x128, 8 waves (2x4), wave tile 64x32 (4x2 WMMA tiles), K-step 32.
// Double-buffered LDS staging via GLOBAL_LOAD_ASYNC_TO_LDS_B128 (ASYNCcnt).
#define TILE_M 128
#define TILE_N 128
#define TILE_K 32

__global__ __launch_bounds__(256) void gemm_bf16_wmma(
    const unsigned short* __restrict__ A,   // [M,K] bf16 bits
    const unsigned short* __restrict__ W,   // [N,K] bf16 bits
    const float* __restrict__ bias,         // [N]
    float* __restrict__ C,                  // [M,N]
    int M, int N, int K, int relu)
{
    __shared__ unsigned short As[2][TILE_M * TILE_K];  // 2 x 8 KB
    __shared__ unsigned short Ws[2][TILE_N * TILE_K];  // 2 x 8 KB

    const int tid  = threadIdx.x;
    const int lane = tid & 31;
    const int wave = tid >> 5;             // 0..7
    const int wm   = (wave >> 2) * 64;     // 0 / 64
    const int wn   = (wave & 3) * 32;      // 0,32,64,96
    const int bm   = blockIdx.y * TILE_M;
    const int bn   = blockIdx.x * TILE_N;
    const int mloc  = lane & 15;
    const int khalf = (lane >> 4) * 8;     // ISA 16-bit A layout: lanes 0-15 -> K 0-7/16-23,
                                           //                      lanes 16-31 -> K 8-15/24-31

    // per-thread staging chunk coordinates (2 chunks of 16B per matrix per stage)
    const int r0 = tid >> 2;               // rows 0..63   (it=0)
    const int r1 = (tid + 256) >> 2;       // rows 64..127 (it=1)
    const int c0 = (tid & 3) << 3;         // col 0,8,16,24

    f32x8 acc[4][2];
#pragma unroll
    for (int i = 0; i < 4; ++i)
#pragma unroll
        for (int j = 0; j < 2; ++j) acc[i][j] = (f32x8)(0.0f);

    const int nw0 = (bn + r0 < N) ? (bn + r0) : (N - 1);   // clamp ragged W rows:
    const int nw1 = (bn + r1 < N) ? (bn + r1) : (N - 1);   // keeps async-count exact

    const int nk = K / TILE_K;

#if USE_ASYNC_LDS
    // issue exactly 4 async-to-LDS instructions per wave per stage
    auto stage = [&](int k0, int buf) {
        async_copy16(&A[(size_t)(bm + r0) * K + k0 + c0], &As[buf][r0 * TILE_K + c0]);
        async_copy16(&W[(size_t)nw0 * K + k0 + c0],       &Ws[buf][r0 * TILE_K + c0]);
        async_copy16(&A[(size_t)(bm + r1) * K + k0 + c0], &As[buf][r1 * TILE_K + c0]);
        async_copy16(&W[(size_t)nw1 * K + k0 + c0],       &Ws[buf][r1 * TILE_K + c0]);
    };
    stage(0, 0);
#endif

    for (int kt = 0; kt < nk; ++kt) {
        const int cur = kt & 1;
#if USE_ASYNC_LDS
        if (kt + 1 < nk) {
            stage((kt + 1) * TILE_K, cur ^ 1);     // prefetch next tile into other buffer
            __builtin_amdgcn_s_wait_asynccnt(4);   // wait for current tile only
        } else {
            __builtin_amdgcn_s_wait_asynccnt(0);
        }
        __syncthreads();                           // make LDS writes visible to all waves
#else
        {
            const int k0 = kt * TILE_K;
            *(uint4*)&As[cur][r0 * TILE_K + c0] =
                *(const uint4*)&A[(size_t)(bm + r0) * K + k0 + c0];
            *(uint4*)&Ws[cur][r0 * TILE_K + c0] =
                *(const uint4*)&W[(size_t)nw0 * K + k0 + c0];
            *(uint4*)&As[cur][r1 * TILE_K + c0] =
                *(const uint4*)&A[(size_t)(bm + r1) * K + k0 + c0];
            *(uint4*)&Ws[cur][r1 * TILE_K + c0] =
                *(const uint4*)&W[(size_t)nw1 * K + k0 + c0];
            if (k0 + TILE_K < K) {
                __builtin_prefetch(&A[(size_t)(bm + r0) * K + k0 + TILE_K + c0], 0, 1);
                __builtin_prefetch(&W[(size_t)nw0 * K + k0 + TILE_K + c0], 0, 1);
            }
        }
        __syncthreads();
#endif

        bf16x16 bfrag[2];
#pragma unroll
        for (int j = 0; j < 2; ++j) {
            const unsigned short* p = &Ws[cur][(wn + j * 16 + mloc) * TILE_K];
            bf16x8 lo = *(const bf16x8*)(p + khalf);
            bf16x8 hi = *(const bf16x8*)(p + 16 + khalf);
#pragma unroll
            for (int e = 0; e < 8; ++e) { bfrag[j][e] = lo[e]; bfrag[j][e + 8] = hi[e]; }
        }
#pragma unroll
        for (int i = 0; i < 4; ++i) {
            const unsigned short* p = &As[cur][(wm + i * 16 + mloc) * TILE_K];
            bf16x8 lo = *(const bf16x8*)(p + khalf);
            bf16x8 hi = *(const bf16x8*)(p + 16 + khalf);
            bf16x16 afrag;
#pragma unroll
            for (int e = 0; e < 8; ++e) { afrag[e] = lo[e]; afrag[e + 8] = hi[e]; }
#pragma unroll
            for (int j = 0; j < 2; ++j) {
                acc[i][j] = __builtin_amdgcn_wmma_f32_16x16x32_bf16(
                    false, afrag, false, bfrag[j], (short)0, acc[i][j], false, false);
            }
        }
        __syncthreads();   // all waves done with buf[cur] before it is re-staged
    }

    // C/D layout: lane = N%16 (+16 for M-high-half); VGPR r -> M = r + 8*(lane>=16)
    const int mrow0 = (lane >> 4) * 8;
#pragma unroll
    for (int i = 0; i < 4; ++i) {
#pragma unroll
        for (int j = 0; j < 2; ++j) {
            const int n = bn + wn + j * 16 + mloc;
            if (n >= N) continue;
            const float bv = bias ? bias[n] : 0.0f;
#pragma unroll
            for (int r = 0; r < 8; ++r) {
                const int m = bm + wm + i * 16 + mrow0 + r;
                float val = acc[i][j][r] + bv;
                if (relu) val = fmaxf(val, 0.0f);
                C[(size_t)m * N + n] = val;
            }
        }
    }
}

// ---------------- attention scores + softmax (one block per (b,h,t)) ----------------
__global__ __launch_bounds__(256) void attn_probs_kernel(
    const float* __restrict__ q, int ldq,
    const float* __restrict__ k, int ldk,
    float* __restrict__ probs, int slen)
{
    const int t = blockIdx.x, h = blockIdx.y, b = blockIdx.z;
    const int tid = threadIdx.x;
    __shared__ float qs[HD];
    __shared__ float red[256];

    const float* qrow = q + (size_t)(b * T + t) * ldq + h * HD;
    for (int i = tid; i < HD; i += 256) qs[i] = qrow[i];
    __syncthreads();

    const int nch = slen >> 8;             // 1 (slen=256) or 2 (slen=512)
    float sc[2];
    for (int c = 0; c < nch; ++c) {
        const int s = c * 256 + tid;
        const float* kr = k + (size_t)(b * slen + s) * ldk + h * HD;
        float d = 0.0f;
        for (int e = 0; e < HD; ++e) d += qs[e] * kr[e];
        sc[c] = d * ATT_SCALE;
    }
    float mx = sc[0];
    if (nch == 2) mx = fmaxf(mx, sc[1]);
    red[tid] = mx; __syncthreads();
    for (int o = 128; o > 0; o >>= 1) {
        if (tid < o) red[tid] = fmaxf(red[tid], red[tid + o]);
        __syncthreads();
    }
    mx = red[0]; __syncthreads();

    float ex[2]; float sum = 0.0f;
    for (int c = 0; c < nch; ++c) { ex[c] = __expf(sc[c] - mx); sum += ex[c]; }
    red[tid] = sum; __syncthreads();
    for (int o = 128; o > 0; o >>= 1) {
        if (tid < o) red[tid] += red[tid + o];
        __syncthreads();
    }
    const float inv = 1.0f / red[0];
    float* pr = probs + (size_t)((b * H + h) * T + t) * slen;
    for (int c = 0; c < nch; ++c) pr[c * 256 + tid] = ex[c] * inv;
}

// ---------------- attention AV: out[b,t,h*HD+e] = sum_s p * v ----------------
__global__ __launch_bounds__(256) void attn_av_kernel(
    const float* __restrict__ probs, const float* __restrict__ v, int ldv,
    float* __restrict__ out, int slen)
{
    const int row = blockIdx.x;            // b*T + t
    const int b = row / T, t = row % T;
    for (int d = threadIdx.x; d < D; d += 256) {
        const int h = d / HD, e = d % HD;
        const float* pr = probs + (size_t)((b * H + h) * T + t) * slen;
        const float* vc = v + (size_t)(b * slen) * ldv + h * HD + e;
        float s = 0.0f;
        for (int si = 0; si < slen; ++si) s += pr[si] * vc[(size_t)si * ldv];
        out[(size_t)row * D + d] = s;
    }
}

// ---------------- residual add + LayerNorm (in-place safe) ----------------
__global__ __launch_bounds__(256) void add_ln_kernel(
    const float* __restrict__ x, const float* __restrict__ y,
    const float* __restrict__ w, const float* __restrict__ b,
    float* __restrict__ out)
{
    const int row = blockIdx.x;
    const int tid = threadIdx.x;
    __shared__ float red[256];
    float vals[9];

    float s = 0.0f; int cnt = 0;
    for (int i = tid; i < D; i += 256) {
        const float vv = x[(size_t)row * D + i] + y[(size_t)row * D + i];
        vals[cnt++] = vv; s += vv;
    }
    red[tid] = s; __syncthreads();
    for (int o = 128; o > 0; o >>= 1) { if (tid < o) red[tid] += red[tid + o]; __syncthreads(); }
    const float mu = red[0] / (float)D; __syncthreads();

    s = 0.0f; cnt = 0;
    for (int i = tid; i < D; i += 256) { const float dv = vals[cnt++] - mu; s += dv * dv; }
    red[tid] = s; __syncthreads();
    for (int o = 128; o > 0; o >>= 1) { if (tid < o) red[tid] += red[tid + o]; __syncthreads(); }
    const float rstd = rsqrtf(red[0] / (float)D + 1e-5f);

    cnt = 0;
    for (int i = tid; i < D; i += 256)
        out[(size_t)row * D + i] = (vals[cnt++] - mu) * rstd * w[i] + b[i];
}

// ---------------- host driver ----------------
extern "C" void kernel_launch(void* const* d_in, const int* in_sizes, int n_in,
                              void* d_out, int out_size, void* d_ws, size_t ws_size,
                              hipStream_t stream)
{
    (void)in_sizes; (void)n_in; (void)out_size; (void)ws_size;

    const int*   tok     = (const int*)d_in[0];
    const float* memf    = (const float*)d_in[1];
    const float* embw    = (const float*)d_in[2];
    const float* posw    = (const float*)d_in[3];
    const float* sa_in_w = (const float*)d_in[4];
    const float* sa_in_b = (const float*)d_in[5];
    const float* sa_ou_w = (const float*)d_in[6];
    const float* sa_ou_b = (const float*)d_in[7];
    const float* ca_in_w = (const float*)d_in[8];
    const float* ca_in_b = (const float*)d_in[9];
    const float* ca_ou_w = (const float*)d_in[10];
    const float* ca_ou_b = (const float*)d_in[11];
    const float* ln1w = (const float*)d_in[12];
    const float* ln1b = (const float*)d_in[13];
    const float* ln2w = (const float*)d_in[14];
    const float* ln2b = (const float*)d_in[15];
    const float* ln3w = (const float*)d_in[16];
    const float* ln3b = (const float*)d_in[17];
    const float* fw1 = (const float*)d_in[18];
    const float* fb1 = (const float*)d_in[19];
    const float* fw2 = (const float*)d_in[20];
    const float* fb2 = (const float*)d_in[21];
    const float* outw = (const float*)d_in[22];
    const float* outb = (const float*)d_in[23];
    float* logits = (float*)d_out;

    // workspace layout (all 256B aligned)
    char* ws = (char*)d_ws;
    size_t off = 0;
    auto walloc = [&](size_t bytes) -> char* {
        char* p = ws + off;
        off += (bytes + 255) & ~(size_t)255;
        return p;
    };
    float* xbuf  = (float*)walloc((size_t)BT * D * 4);
    float* ybuf  = (float*)walloc((size_t)BT * D * 4);
    float* qkv   = (float*)walloc(((size_t)BT * D + (size_t)BS * 2 * D) * 4);
    float* hbuf  = (float*)walloc((size_t)BT * F * 4);
    float* probs = (float*)walloc((size_t)B * H * T * S * 4);
    unsigned short* abf   = (unsigned short*)walloc((size_t)BT * F * 2);
    unsigned short* membf = (unsigned short*)walloc((size_t)BS * D * 2);
    unsigned short* wbf   = (unsigned short*)walloc((size_t)F * D * 2);

    auto cvt = [&](const float* src, unsigned short* dst, size_t n) {
        size_t blocks = (n + 255) / 256;
        if (blocks > 4096) blocks = 4096;
        cvt_bf16_kernel<<<(int)blocks, 256, 0, stream>>>(src, dst, n);
    };
    auto gemm = [&](const unsigned short* A, const unsigned short* Wm, const float* bias,
                    float* C, int M, int N, int K, int relu) {
        dim3 grid((N + TILE_N - 1) / TILE_N, M / TILE_M);
        gemm_bf16_wmma<<<grid, 256, 0, stream>>>(A, Wm, bias, C, M, N, K, relu);
    };

    embed_kernel<<<BT, 256, 0, stream>>>(tok, embw, posw, xbuf);
    cvt(memf, membf, (size_t)BS * D);

    for (int l = 0; l < L; ++l) {
        const float* siw = sa_in_w + (size_t)l * 3 * D * D;
        const float* sib = sa_in_b + (size_t)l * 3 * D;
        const float* sow = sa_ou_w + (size_t)l * D * D;
        const float* sob = sa_ou_b + (size_t)l * D;
        const float* ciw = ca_in_w + (size_t)l * 3 * D * D;
        const float* cib = ca_in_b + (size_t)l * 3 * D;
        const float* cow = ca_ou_w + (size_t)l * D * D;
        const float* cob = ca_ou_b + (size_t)l * D;
        const float* w1 = fw1 + (size_t)l * F * D;
        const float* b1 = fb1 + (size_t)l * F;
        const float* w2 = fw2 + (size_t)l * D * F;
        const float* b2 = fb2 + (size_t)l * D;

        // ---- self-attention ----
        cvt(xbuf, abf, (size_t)BT * D);
        cvt(siw, wbf, (size_t)3 * D * D);
        gemm(abf, wbf, sib, qkv, BT, 3 * D, D, 0);
        {
            dim3 g(T, H, B);
            attn_probs_kernel<<<g, 256, 0, stream>>>(qkv, 3 * D, qkv + D, 3 * D, probs, T);
        }
        attn_av_kernel<<<BT, 256, 0, stream>>>(probs, qkv + 2 * D, 3 * D, hbuf, T);
        cvt(hbuf, abf, (size_t)BT * D);
        cvt(sow, wbf, (size_t)D * D);
        gemm(abf, wbf, sob, ybuf, BT, D, D, 0);
        add_ln_kernel<<<BT, 256, 0, stream>>>(xbuf, ybuf,
                                              ln1w + (size_t)l * D, ln1b + (size_t)l * D, xbuf);

        // ---- cross-attention ----
        float* qbuf  = qkv;
        float* kvbuf = qkv + (size_t)BT * D;
        cvt(xbuf, abf, (size_t)BT * D);
        cvt(ciw, wbf, (size_t)D * D);                       // q rows of in_proj
        gemm(abf, wbf, cib, qbuf, BT, D, D, 0);
        cvt(ciw + (size_t)D * D, wbf, (size_t)2 * D * D);   // k,v rows
        gemm(membf, wbf, cib + D, kvbuf, BS, 2 * D, D, 0);
        {
            dim3 g(T, H, B);
            attn_probs_kernel<<<g, 256, 0, stream>>>(qbuf, D, kvbuf, 2 * D, probs, S);
        }
        attn_av_kernel<<<BT, 256, 0, stream>>>(probs, kvbuf + D, 2 * D, hbuf, S);
        cvt(hbuf, abf, (size_t)BT * D);
        cvt(cow, wbf, (size_t)D * D);
        gemm(abf, wbf, cob, ybuf, BT, D, D, 0);
        add_ln_kernel<<<BT, 256, 0, stream>>>(xbuf, ybuf,
                                              ln2w + (size_t)l * D, ln2b + (size_t)l * D, xbuf);

        // ---- FFN ----
        cvt(xbuf, abf, (size_t)BT * D);
        cvt(w1, wbf, (size_t)F * D);
        gemm(abf, wbf, b1, hbuf, BT, F, D, 1);              // fused ReLU
        cvt(hbuf, abf, (size_t)BT * F);
        cvt(w2, wbf, (size_t)D * F);
        gemm(abf, wbf, b2, ybuf, BT, D, F, 0);
        add_ln_kernel<<<BT, 256, 0, stream>>>(xbuf, ybuf,
                                              ln3w + (size_t)l * D, ln3b + (size_t)l * D, xbuf);
    }

    // ---- output projection (N=1000, ragged tile guarded) ----
    cvt(xbuf, abf, (size_t)BT * D);
    cvt(outw, wbf, (size_t)V * D);
    gemm(abf, wbf, outb, logits, BT, V, D, 0);
}
